// Decoder_70042326663970
// MI455X (gfx1250) — matrix operand order repaired and checked by reading, not compiled, not run on hardware
//
#include <hip/hip_runtime.h>
#include <math.h>

#define HID 128
#define EMB 128
#define SRC 128
#define BATCH 512
#define TSTEPS 254
#define INIT_CAP 20.0f

typedef __attribute__((ext_vector_type(16))) _Float16 v16h;
typedef __attribute__((ext_vector_type(8)))  _Float16 v8h;
typedef __attribute__((ext_vector_type(8)))  float    v8f;

union V16 { v16h v; struct { v8h lo, hi; } s; };

__device__ __forceinline__ float fast_tanh(float x) {
#if __has_builtin(__builtin_amdgcn_tanhf)
  return __builtin_amdgcn_tanhf(x);      // v_tanh_f32 on gfx1250
#else
  return tanhf(x);
#endif
}
__device__ __forceinline__ float sigm(float x) { return 1.0f / (1.0f + __expf(-x)); }

__device__ __forceinline__ unsigned mix3(unsigned a, unsigned b, unsigned c) {
  unsigned h = a * 0x9E3779B1u ^ b * 0x85EBCA77u ^ c * 0xC2B2AE3Du;
  h ^= h >> 16; h *= 0x7FEB352Du; h ^= h >> 15; h *= 0x846CA68Bu; h ^= h >> 16;
  return h;
}

// ---------------------------------------------------------------------------
// Prep 1: pack [Wi|Wh] (512x256) and Wq (128x128) into f16 workspace
// ---------------------------------------------------------------------------
__global__ void prep_weights(const float* __restrict__ Wi, const float* __restrict__ Wh,
                             const float* __restrict__ Wq, _Float16* __restrict__ Wcat,
                             _Float16* __restrict__ Wqh) {
  int n = blockIdx.x * blockDim.x + threadIdx.x;
  int stride = gridDim.x * blockDim.x;
  for (int i = n; i < 512 * 256; i += stride) {
    int g = i >> 8, k = i & 255;
    float w = (k < 128) ? Wi[g * 128 + k] : Wh[g * 128 + (k - 128)];
    Wcat[i] = (_Float16)w;
  }
  for (int i = n; i < 128 * 128; i += stride) Wqh[i] = (_Float16)Wq[i];
}

// ---------------------------------------------------------------------------
// Prep 2: e[b][o][s] = sum_h Wr[o][h]*context[s][b][h] + br[o]  (WMMA)
//   layout chosen [b][h(=o)][s] so the decode tanh-scan reads contiguous s.
// ---------------------------------------------------------------------------
__global__ __launch_bounds__(256)
void prep_e(const float* __restrict__ context, const float* __restrict__ Wr,
            const float* __restrict__ br, float* __restrict__ e_ws) {
  __shared__ _Float16 A[16][128];
  const int tid = threadIdx.x;
  const int lane = tid & 31;
  const int wid = tid >> 5;
  const int bt = blockIdx.x >> 7;
  const int s  = blockIdx.x & 127;
  const int b0 = bt * 16;

  const float* ctx = context + (size_t)s * (BATCH * HID);
  for (int i = tid; i < 16 * 128; i += 256) {
    int r = i >> 7, h = i & 127;
    A[r][h] = (_Float16)ctx[(size_t)(b0 + r) * HID + h];
  }
  __syncthreads();

  const int mrow  = lane & 15;
  const int abase = (lane >> 4) * 16;
  const int bkoff = (lane >> 4) * 16;
  const int otile = wid;               // 8 waves -> 8 tiles of 16 -> o = 0..127
  const v8f vzero = {0.f,0.f,0.f,0.f,0.f,0.f,0.f,0.f};
  v8f acc = vzero;

  #pragma unroll
  for (int ks = 0; ks < 4; ++ks) {
    const int a0 = ks * 32 + abase;
    V16 a;
    a.s.lo = *(const v8h*)&A[mrow][a0];
    a.s.hi = *(const v8h*)&A[mrow][a0 + 8];

    const int g  = otile * 16 + (lane & 15);
    const int k0 = ks * 32 + bkoff;
    const float4* wp = (const float4*)(Wr + (size_t)g * HID + k0);
    float4 w0 = wp[0], w1 = wp[1], w2 = wp[2], w3 = wp[3];
    float wf[16] = {w0.x,w0.y,w0.z,w0.w, w1.x,w1.y,w1.z,w1.w,
                    w2.x,w2.y,w2.z,w2.w, w3.x,w3.y,w3.z,w3.w};
    V16 bm;
    #pragma unroll
    for (int i = 0; i < 16; ++i) bm.v[i] = (_Float16)wf[i];

    acc = __builtin_amdgcn_wmma_f32_16x16x32_f16(false, a.v, false, bm.v,
                                                 (short)0, acc, false, false);
  }

  #pragma unroll
  for (int r = 0; r < 8; ++r) {
    int m = (lane >> 4) * 8 + r;
    int n = lane & 15;
    int o = otile * 16 + n;
    e_ws[(size_t)(b0 + m) * (HID * SRC) + (size_t)o * SRC + s] = acc[r] + br[o];
  }
}

// ---------------------------------------------------------------------------
// Persistent decode: 32 WGs x 256 threads; each WG owns 16 batch rows for all
// 254 steps. 8 waves share the WMMA GEMMs; 16-thread teams own one row each
// for attention / softmax / sampling / vehicle-state.
// ---------------------------------------------------------------------------
__global__ __launch_bounds__(256)
void vrp_decode(const float* __restrict__ dec0, const float* __restrict__ bbe,
                const float* __restrict__ emb,  const float* __restrict__ hx,
                const float* __restrict__ cx,   const float* __restrict__ bi,
                const float* __restrict__ bh,   const float* __restrict__ bq,
                const float* __restrict__ vvec, const float* __restrict__ e_ws,
                const _Float16* __restrict__ Wcat, const _Float16* __restrict__ Wqh,
                float* __restrict__ out) {
  __shared__ _Float16 A_lds[16][256];     // [dec_in | h] in f16
  __shared__ float gates_lds[16][512];
  __shared__ float q_lds[16][128];
  __shared__ float c_lds[16][128];
  __shared__ float bcat_lds[512];
  __shared__ float bq_lds[128];
  __shared__ float v_lds[128];

  const int tid  = threadIdx.x;
  const int lane = tid & 31;
  const int wid  = tid >> 5;
  const int b0   = blockIdx.x * 16;
  const int team = tid >> 4;             // row 0..15
  const int tt   = tid & 15;
  const int b    = b0 + team;

  float* out_probs = out;
  float* out_sel   = out_probs + (size_t)TSTEPS * BATCH * SRC;
  float* out_h     = out_sel   + (size_t)TSTEPS * BATCH;
  float* out_c     = out_h     + (size_t)BATCH * HID;
  float* out_state = out_c     + (size_t)BATCH * HID;

  for (int i = tid; i < 512; i += 256) bcat_lds[i] = bi[i] + bh[i];
  for (int i = tid; i < 128; i += 256) { bq_lds[i] = bq[i]; v_lds[i] = vvec[i]; }
  for (int i = tid; i < 16 * 128; i += 256) {
    int r = i >> 7, h = i & 127;
    A_lds[r][h]       = (_Float16)dec0[(size_t)(b0 + r) * EMB + h];
    A_lds[r][128 + h] = (_Float16)hx  [(size_t)(b0 + r) * HID + h];
    c_lds[r][h]       = cx[(size_t)(b0 + r) * HID + h];
  }
  // per-row carried state, maintained redundantly by all 16 team threads
  unsigned mw0 = 0, mw1 = 0, mw2 = 0, mw3 = 0;
  int   cur_idx = 0;
  float rc = INIT_CAP, ctm = 0.f, dtot = 0.f, pcv = 0.f, ptv = 0.f;
  __syncthreads();

  const int mrow  = lane & 15;
  const int abase = (lane >> 4) * 16;
  const int bkoff = (lane >> 4) * 16;
  const float* ebase = e_ws + (size_t)b * (HID * SRC);
  const v8f vzero = {0.f,0.f,0.f,0.f,0.f,0.f,0.f,0.f};

  #pragma unroll 1
  for (int t = 0; t < TSTEPS; ++t) {
    // ---- phase 1: gates = [dec_in|h] @ Wcat^T  (16x512, K=256) ----
    v8f acc[4];
    #pragma unroll
    for (int nt = 0; nt < 4; ++nt) acc[nt] = vzero;
    #pragma unroll
    for (int ks = 0; ks < 8; ++ks) {
      const int a0 = ks * 32 + abase;
      V16 a;
      a.s.lo = *(const v8h*)&A_lds[mrow][a0];
      a.s.hi = *(const v8h*)&A_lds[mrow][a0 + 8];
      const int bk0 = ks * 32 + bkoff;
      #pragma unroll
      for (int nt = 0; nt < 4; ++nt) {
        const int g = (wid * 4 + nt) * 16 + (lane & 15);
        const v8h* bp = (const v8h*)(Wcat + (size_t)g * 256 + bk0);
        V16 bm; bm.s.lo = bp[0]; bm.s.hi = bp[1];
        acc[nt] = __builtin_amdgcn_wmma_f32_16x16x32_f16(false, a.v, false, bm.v,
                                                         (short)0, acc[nt], false, false);
      }
    }
    #pragma unroll
    for (int nt = 0; nt < 4; ++nt) {
      #pragma unroll
      for (int r = 0; r < 8; ++r)
        gates_lds[(lane >> 4) * 8 + r][(wid * 4 + nt) * 16 + (lane & 15)] = acc[nt][r];
    }
    __syncthreads();

    // ---- phase 2: LSTM pointwise ----
    #pragma unroll
    for (int j = 0; j < 8; ++j) {
      int idx = tid + j * 256;
      int r = idx >> 7, h = idx & 127;
      float ig = gates_lds[r][h]       + bcat_lds[h];
      float fg = gates_lds[r][128 + h] + bcat_lds[128 + h];
      float gg = gates_lds[r][256 + h] + bcat_lds[256 + h];
      float og = gates_lds[r][384 + h] + bcat_lds[384 + h];
      float cp = c_lds[r][h];
      float cy = sigm(fg) * cp + sigm(ig) * fast_tanh(gg);
      float hy = sigm(og) * fast_tanh(cy);
      c_lds[r][h] = cy;
      A_lds[r][128 + h] = (_Float16)hy;
    }
    __syncthreads();

    // ---- phase 3: q = hy @ Wq^T + bq  (16x128, K=128) ----
    {
      v8f qa = vzero;
      #pragma unroll
      for (int ks = 0; ks < 4; ++ks) {
        const int a0 = 128 + ks * 32 + abase;
        V16 a;
        a.s.lo = *(const v8h*)&A_lds[mrow][a0];
        a.s.hi = *(const v8h*)&A_lds[mrow][a0 + 8];
        const int g  = wid * 16 + (lane & 15);
        const int k0 = ks * 32 + bkoff;
        const v8h* bp = (const v8h*)(Wqh + (size_t)g * 128 + k0);
        V16 bm; bm.s.lo = bp[0]; bm.s.hi = bp[1];
        qa = __builtin_amdgcn_wmma_f32_16x16x32_f16(false, a.v, false, bm.v,
                                                    (short)0, qa, false, false);
      }
      #pragma unroll
      for (int r = 0; r < 8; ++r) {
        int m = (lane >> 4) * 8 + r;
        int o = wid * 16 + (lane & 15);
        q_lds[m][o] = qa[r] + bq_lds[o];
      }
    }
    __syncthreads();

    // ---- phase 4: u[b][s] = sum_h v[h]*tanh(q[b][h]+e[b][h][s]) ----
    float uacc[8];
    #pragma unroll
    for (int k = 0; k < 8; ++k) uacc[k] = 0.f;
    #pragma unroll 4
    for (int h = 0; h < 128; ++h) {
      float qv = q_lds[team][h];
      float vv = v_lds[h];
      const float* ep = ebase + (size_t)h * SRC + tt;
      __builtin_prefetch(ep + 4 * SRC, 0, 0);   // global_prefetch_b8 on the e stream
      #pragma unroll
      for (int k = 0; k < 8; ++k)
        uacc[k] += vv * fast_tanh(qv + ep[16 * k]);
    }

    // ---- phase 5: mask update (pure fn of carried state; no barrier) ----
    {
      unsigned bitw = 1u << (cur_idx & 31);
      if      (cur_idx < 32) mw0 |= bitw;
      else if (cur_idx < 64) mw1 |= bitw;
      else if (cur_idx < 96) mw2 |= bitw;
      else                   mw3 |= bitw;
      unsigned unv = ((~mw0) & 0xFFFFFFFEu) | (~mw1) | (~mw2) | (~mw3);
      bool depot_forced = (cur_idx == 0) && (unv != 0u);
      mw0 = (mw0 & ~1u) | (depot_forced ? 1u : 0u);
    }

    // ---- phase 6: softmax (team reduce), probs out, Gumbel-max sample ----
    float lg[8];
    float mx = -INFINITY;
    #pragma unroll
    for (int k = 0; k < 8; ++k) {
      int s = tt + 16 * k;
      unsigned w = (s < 64) ? ((s < 32) ? mw0 : mw1) : ((s < 96) ? mw2 : mw3);
      bool msk = (w >> (s & 31)) & 1u;
      lg[k] = msk ? -INFINITY : uacc[k];
      mx = fmaxf(mx, lg[k]);
    }
    #pragma unroll
    for (int off = 8; off >= 1; off >>= 1) mx = fmaxf(mx, __shfl_xor(mx, off, 16));
    float sume = 0.f;
    #pragma unroll
    for (int k = 0; k < 8; ++k) sume += __expf(lg[k] - mx);
    #pragma unroll
    for (int off = 8; off >= 1; off >>= 1) sume += __shfl_xor(sume, off, 16);
    float inv = 1.0f / sume;

    float* pout = out_probs + (size_t)t * (BATCH * SRC) + (size_t)b * SRC;
    float bv = -INFINITY; int bix = 0;
    #pragma unroll
    for (int k = 0; k < 8; ++k) {
      int s = tt + 16 * k;
      pout[s] = __expf(lg[k] - mx) * inv;
      unsigned hsh = mix3((unsigned)t, (unsigned)b, (unsigned)s);
      float u01 = ((float)(hsh >> 8) + 0.5f) * (1.0f / 16777216.0f);
      float gum = -__logf(-__logf(u01));
      float sc = lg[k] + gum;
      if (sc > bv || (sc == bv && s < bix)) { bv = sc; bix = s; }
    }
    #pragma unroll
    for (int off = 8; off >= 1; off >>= 1) {
      float ov = __shfl_xor(bv, off, 16);
      int   oi = __shfl_xor(bix, off, 16);
      if (ov > bv || (ov == bv && oi < bix)) { bv = ov; bix = oi; }
    }
    const int new_idx = bix;   // identical across team

    // ---- phase 7: vehicle state (redundant per team; writes gated) ----
    {
      const float* po = bbe + (size_t)cur_idx * (BATCH * 5) + (size_t)b * 5;
      const float* pn = bbe + (size_t)new_idx * (BATCH * 5) + (size_t)b * 5;
      float dx = po[0] - pn[0], dy = po[1] - pn[1];
      float sq = dx * dx + dy * dy;
      float dist = (sq > 0.f) ? sqrtf(sq) : 0.f;
      if (new_idx != 0) {
        float ctn = fmaxf(ctm + dist, pn[3]);
        float rcn = rc - pn[2];
        ptv += fmaxf(ctn - pn[4], 0.f);
        pcv += fmaxf(-rcn, 0.f);
        rc = rcn; ctm = ctn;
      } else {
        rc = INIT_CAP; ctm = 0.f;
      }
      dtot += dist;
      cur_idx = new_idx;
      if (tt == 0) out_sel[(size_t)t * BATCH + b] = (float)new_idx;
    }

    // ---- phase 8: next decoder input = embedded_inputs[new_idx][b][:] ----
    {
      const float* ev = emb + (size_t)new_idx * (BATCH * EMB) + (size_t)b * EMB;
      #pragma unroll
      for (int k = 0; k < 8; ++k) {
        int h = tt + 16 * k;
        A_lds[team][h] = (_Float16)ev[h];
      }
    }
    __syncthreads();
  }

  // ---- final outputs: hF, cF, state[:, 2:5] ----
  #pragma unroll
  for (int k = 0; k < 8; ++k) {
    int h = tt + 16 * k;
    out_h[(size_t)b * HID + h] = (float)A_lds[team][128 + h];
    out_c[(size_t)b * HID + h] = c_lds[team][h];
  }
  if (tt == 0) {
    out_state[(size_t)b * 3 + 0] = dtot;
    out_state[(size_t)b * 3 + 1] = pcv;
    out_state[(size_t)b * 3 + 2] = ptv;
  }
}

// ---------------------------------------------------------------------------
extern "C" void kernel_launch(void* const* d_in, const int* in_sizes, int n_in,
                              void* d_out, int out_size, void* d_ws, size_t ws_size,
                              hipStream_t stream) {
  (void)in_sizes; (void)n_in; (void)out_size; (void)ws_size;
  const float* dec0 = (const float*)d_in[0];
  const float* bbe  = (const float*)d_in[1];
  const float* emb  = (const float*)d_in[2];
  const float* hx   = (const float*)d_in[3];
  const float* cx   = (const float*)d_in[4];
  const float* ctx  = (const float*)d_in[5];
  const float* Wi   = (const float*)d_in[6];
  const float* bi   = (const float*)d_in[7];
  const float* Wh   = (const float*)d_in[8];
  const float* bh   = (const float*)d_in[9];
  const float* Wq   = (const float*)d_in[10];
  const float* bq   = (const float*)d_in[11];
  const float* Wr   = (const float*)d_in[12];
  const float* br   = (const float*)d_in[13];
  const float* vv   = (const float*)d_in[14];

  float*    e_ws = (float*)d_ws;                                     // 33.5 MB
  _Float16* Wcat = (_Float16*)((char*)d_ws + (size_t)BATCH * HID * SRC * 4);
  _Float16* Wqh  = (_Float16*)((char*)Wcat + (size_t)512 * 256 * 2);

  prep_weights<<<256, 256, 0, stream>>>(Wi, Wh, Wq, Wcat, Wqh);
  prep_e<<<dim3((BATCH / 16) * SRC), 256, 0, stream>>>(ctx, Wr, br, e_ws);
  vrp_decode<<<dim3(BATCH / 16), 256, 0, stream>>>(dec0, bbe, emb, hx, cx,
                                                   bi, bh, bq, vv,
                                                   e_ws, Wcat, Wqh, (float*)d_out);
}
